// WindowAttention_72688026517936
// MI455X (gfx1250) — matrix operand order repaired and checked by reading, not compiled, not run on hardware
//
#include <hip/hip_runtime.h>
#include <hip/hip_bf16.h>

// ---------------------------------------------------------------------------
// WindowAttention for MI455X (gfx1250, wave32, WMMA).
// B=2048 windows, N=64 tokens, DIM=256, 1 head. All GEMMs use
// v_wmma_f32_16x16x32_bf16 with fp32 accumulation; softmax in fp32.
// LDS = 155,648 B per workgroup -> 2 workgroups (16 wave32) per 320 KB WGP.
// ---------------------------------------------------------------------------

typedef __attribute__((ext_vector_type(16))) __bf16 bf16x16;
typedef __attribute__((ext_vector_type(8)))  __bf16 bf16x8;
typedef __attribute__((ext_vector_type(4)))  __bf16 bf16x4;
typedef __attribute__((ext_vector_type(8)))  float  v8f;

#define NTOK   64          // tokens per window
#define DIM    256
#define LDX    264         // bf16 row stride for 256-wide LDS buffers (pad 8)
#define LDVT   72          // bf16 row stride for V^T buffer (64 + 8)
#define LDS_S  68          // f32 row stride for score buffer (64 + 4)
#define LDP    (2*LDS_S)   // bf16 row stride of in-place P rows (136)

// LDS pool offsets (bytes)
#define OFF_X   0                         // 64 x 264 bf16  (reused for O)
#define OFF_Q   (OFF_X  + NTOK*LDX*2)     // 64 x 264 bf16
#define OFF_K   (OFF_Q  + NTOK*LDX*2)     // 64 x 264 bf16
#define OFF_VT  (OFF_K  + NTOK*LDX*2)     // 256 x 72 bf16  (V transposed)
#define OFF_S   (OFF_VT + DIM*LDVT*2)     // 64 x 68 f32 (scores; P bf16 in place)
#define SMEM_BYTES (OFF_S + NTOK*LDS_S*4) // = 155648 B  (<= 163840 -> 2 WG/WGP)

// Workspace layout (bytes)
#define WS_WQKV   0                        // 768*256 bf16
#define WS_WPROJ  (768*256*2)              // 256*256 bf16
#define WS_BIAS   (WS_WPROJ + 256*256*2)   // 64*64  f32

// ---------------------------------------------------------------------------
// Fragment loaders (wave32); A and B share this code's own contiguous K order:
//  A 16x32 bf16: M=lane&15, K = (lane>>4)*16 + e  (16 contiguous K per lane)
//  B 32x16 bf16: N=lane&15, K = (lane>>4)*16 + e  (16 contiguous K per lane)
//  C/D 16x16 f32: N=lane&15, row r -> M = r + (lane>>4)*8
// ---------------------------------------------------------------------------
__device__ __forceinline__
bf16x16 load_frag_a(const __bf16* base, int ld, int m0, int k0, int lane) {
    const int m  = m0 + (lane & 15);
    const int kb = k0 + ((lane >> 4) << 4);
    const __bf16* p = base + m * ld + kb;
    bf16x8 lo = *reinterpret_cast<const bf16x8*>(p);       // K = kb .. kb+7
    bf16x8 hi = *reinterpret_cast<const bf16x8*>(p + 8);   // K = kb+8 .. kb+15
    bf16x16 r;
#pragma unroll
    for (int e = 0; e < 8; ++e) { r[e] = lo[e]; r[e + 8] = hi[e]; }
    return r;
}

// Source layout: element (K=k, N=n) lives at base[n*ld + k] (contiguous K).
__device__ __forceinline__
bf16x16 load_frag_b(const __bf16* base, int ld, int n0, int k0, int lane) {
    const int n  = n0 + (lane & 15);
    const int kb = k0 + ((lane >> 4) << 4);
    const __bf16* p = base + n * ld + kb;
    bf16x8 lo = *reinterpret_cast<const bf16x8*>(p);
    bf16x8 hi = *reinterpret_cast<const bf16x8*>(p + 8);
    bf16x16 r;
#pragma unroll
    for (int e = 0; e < 8; ++e) { r[e] = lo[e]; r[e + 8] = hi[e]; }
    return r;
}

// ---------------------------------------------------------------------------
// Prep kernel 1: convert fp32 weights -> bf16 workspace (once per call;
// weights then live in the 192 MB L2 for all 2048 blocks).
// ---------------------------------------------------------------------------
__global__ __launch_bounds__(256)
void prep_weights_kernel(const float* __restrict__ qkv_w,
                         const float* __restrict__ proj_w,
                         __bf16* __restrict__ wqkv,
                         __bf16* __restrict__ wproj) {
    int i = blockIdx.x * 256 + threadIdx.x;          // 0 .. 262143
    if (i < 768 * 256) {
        wqkv[i] = (__bf16)qkv_w[i];
    } else {
        int j = i - 768 * 256;
        wproj[j] = (__bf16)proj_w[j];
    }
}

// ---------------------------------------------------------------------------
// Prep kernel 2: 64x64 relative-position bias table (batch invariant).
// bias[n,m] = a_p[mod(az,15)]*cos(az*2pi/64) + b_p[..]*sin(..)
//           + a_r[mod(rad,15)]*cos(rad*tmax[m]/64) + b_r[..]*sin(..)
// with rad = row(n)-row(m), az = col(n)-col(m), tmax[m] = theta_max[m>>3].
// ---------------------------------------------------------------------------
__global__ __launch_bounds__(256)
void prep_bias_kernel(const float* __restrict__ theta_max,
                      const float* __restrict__ a_p, const float* __restrict__ b_p,
                      const float* __restrict__ a_r, const float* __restrict__ b_r,
                      float* __restrict__ bias) {
    int i = blockIdx.x * 256 + threadIdx.x;          // 0 .. 4095
    int n = i >> 6, m = i & 63;
    int rad = (n >> 3) - (m >> 3);                   // [-7, 7]
    int az  = (n & 7)  - (m & 7);                    // [-7, 7]
    int az_idx = ((az  % 15) + 15) % 15;
    int r_idx  = ((rad % 15) + 15) % 15;
    float az_ang = (float)az * (2.0f * 3.14159265358979323846f / 64.0f);
    float A_phi = a_p[az_idx] * __cosf(az_ang) + b_p[az_idx] * __sinf(az_ang);
    float tmax  = theta_max[m >> 3];
    float r_ang = (float)rad * tmax * (1.0f / 64.0f);
    float A_th  = a_r[r_idx] * __cosf(r_ang) + b_r[r_idx] * __sinf(r_ang);
    bias[n * 64 + m] = A_phi + A_th;
}

// ---------------------------------------------------------------------------
// Main fused attention kernel: one workgroup (8 wave32) per window.
// ---------------------------------------------------------------------------
__global__ __launch_bounds__(256)
void attn_kernel(const float*  __restrict__ x,
                 const __bf16* __restrict__ wqkv,   // [768][256] bf16
                 const float*  __restrict__ bqkv,   // [768]
                 const __bf16* __restrict__ wproj,  // [256][256] bf16
                 const float*  __restrict__ bproj,  // [256]
                 const float*  __restrict__ bias,   // [64][64]
                 float*        __restrict__ out) {
    extern __shared__ char smem[];
    __bf16* xbf  = reinterpret_cast<__bf16*>(smem + OFF_X);   // X, later O
    __bf16* qbf  = reinterpret_cast<__bf16*>(smem + OFF_Q);
    __bf16* kbf  = reinterpret_cast<__bf16*>(smem + OFF_K);
    __bf16* vt   = reinterpret_cast<__bf16*>(smem + OFF_VT);  // V^T [256][72]
    float*  sbuf = reinterpret_cast<float*>(smem + OFF_S);    // scores (f32)
    __bf16* pbuf = reinterpret_cast<__bf16*>(smem + OFF_S);   // P in place (bf16)

    const int tid  = threadIdx.x;
    const int lane = tid & 31;
    const int wv   = tid >> 5;
    const int lo16 = lane & 15;
    const int hi8  = (lane >> 4) << 3;
    const size_t bofs = (size_t)blockIdx.x * (NTOK * DIM);
    const float  SCALE = 0.0625f;        // 256^-0.5

    // ---- Phase 1: X (fp32 global) -> bf16 LDS -----------------------------
    const float4* xg = reinterpret_cast<const float4*>(x + bofs);
#pragma unroll 4
    for (int i = tid; i < (NTOK * DIM / 4); i += 256) {
        float4 v = xg[i];
        int e0 = i << 2;
        int row = e0 >> 8, col = e0 & 255;
        bf16x4 bv;
        bv[0] = (__bf16)v.x; bv[1] = (__bf16)v.y;
        bv[2] = (__bf16)v.z; bv[3] = (__bf16)v.w;
        *reinterpret_cast<bf16x4*>(xbf + row * LDX + col) = bv;
    }
    __syncthreads();

    // ---- Phase 2: QKV = X @ Wqkv^T + b  (192 16x16 tiles, 24 per wave) ----
    for (int t = wv; t < 192; t += 8) {
        const int ct = t % 48, tm = t / 48;      // ct: output-channel tile 0..47
        const int cbase = ct << 4;               // channel base in [0,768)
        v8f acc;
        const float bcol = bqkv[cbase + lo16];
#pragma unroll
        for (int r = 0; r < 8; ++r) acc[r] = bcol;
#pragma unroll
        for (int kc = 0; kc < DIM; kc += 32) {
            bf16x16 a = load_frag_a(xbf, LDX, tm << 4, kc, lane);
            bf16x16 b = load_frag_b(wqkv, DIM, cbase, kc, lane);
            acc = __builtin_amdgcn_wmma_f32_16x16x32_bf16(
                      false, a, false, b, (short)0, acc, false, false);
        }
        const int w = ct >> 4, tn = ct & 15;
        const int mb = (tm << 4) + hi8;
        const int n  = (tn << 4) + lo16;
        if (w == 0) {                            // Q (scale folded in)
#pragma unroll
            for (int r = 0; r < 8; ++r)
                qbf[(mb + r) * LDX + n] = (__bf16)(acc[r] * SCALE);
        } else if (w == 1) {                     // K
#pragma unroll
            for (int r = 0; r < 8; ++r)
                kbf[(mb + r) * LDX + n] = (__bf16)acc[r];
        } else {                                 // V stored transposed
#pragma unroll
            for (int r = 0; r < 8; ++r)
                vt[n * LDVT + (mb + r)] = (__bf16)acc[r];
        }
    }
    __syncthreads();

    // ---- Phase 3: S = (Q*scale) @ K^T + bias  (16 tiles, 2 per wave) ------
    for (int t = wv; t < 16; t += 8) {
        const int tm = t >> 2, tn = t & 3;
        v8f acc;
#pragma unroll
        for (int r = 0; r < 8; ++r)
            acc[r] = bias[((tm << 4) + hi8 + r) * 64 + (tn << 4) + lo16];
#pragma unroll
        for (int kc = 0; kc < DIM; kc += 32) {
            bf16x16 a = load_frag_a(qbf, LDX, tm << 4, kc, lane);
            bf16x16 b = load_frag_b(kbf, LDX, tn << 4, kc, lane);
            acc = __builtin_amdgcn_wmma_f32_16x16x32_bf16(
                      false, a, false, b, (short)0, acc, false, false);
        }
#pragma unroll
        for (int r = 0; r < 8; ++r)
            sbuf[((tm << 4) + hi8 + r) * LDS_S + (tn << 4) + lo16] = acc[r];
    }
    __syncthreads();

    // ---- Phase 4: fp32 row softmax; P written in place as bf16 ------------
    // Thread t owns row t entirely; bf16 write at byte 2j trails the fp32
    // read at byte 4j, so narrowing in place is hazard-free.
    if (tid < NTOK) {
        float*  row  = sbuf + tid * LDS_S;
        __bf16* prow = reinterpret_cast<__bf16*>(row);
        float mx = -3.402823466e38f;
#pragma unroll 8
        for (int j = 0; j < 64; ++j) mx = fmaxf(mx, row[j]);
        float s = 0.f;
#pragma unroll 8
        for (int j = 0; j < 64; ++j) { float e = __expf(row[j] - mx); row[j] = e; s += e; }
        float inv = __frcp_rn(s);
#pragma unroll 8
        for (int j = 0; j < 64; ++j)
            prow[j] = (__bf16)(row[j] * inv);
    }
    __syncthreads();

    // ---- Phase 5: O = P @ V  (64 tiles, 8 per wave) -> xbf (X is dead) ----
    for (int t = wv; t < 64; t += 8) {
        const int tm = t >> 4, tn = t & 15;
        v8f acc;
#pragma unroll
        for (int r = 0; r < 8; ++r) acc[r] = 0.f;
#pragma unroll
        for (int kc = 0; kc < NTOK; kc += 32) {
            bf16x16 a = load_frag_a(pbuf, LDP,  tm << 4, kc, lane);
            bf16x16 b = load_frag_b(vt,   LDVT, tn << 4, kc, lane);
            acc = __builtin_amdgcn_wmma_f32_16x16x32_bf16(
                      false, a, false, b, (short)0, acc, false, false);
        }
        const int mb = (tm << 4) + hi8, n = (tn << 4) + lo16;
#pragma unroll
        for (int r = 0; r < 8; ++r)
            xbf[(mb + r) * LDX + n] = (__bf16)acc[r];
    }
    __syncthreads();

    // ---- Phase 6: Y = O @ Wproj^T + b -> global fp32 ----------------------
    float* og = out + bofs;
    for (int t = wv; t < 64; t += 8) {
        const int tm = t >> 4, tn = t & 15;
        const int cb = tn << 4;
        v8f acc;
        const float bc = bproj[cb + lo16];
#pragma unroll
        for (int r = 0; r < 8; ++r) acc[r] = bc;
#pragma unroll
        for (int kc = 0; kc < DIM; kc += 32) {
            bf16x16 a = load_frag_a(xbf, LDX, tm << 4, kc, lane);
            bf16x16 b = load_frag_b(wproj, DIM, cb, kc, lane);
            acc = __builtin_amdgcn_wmma_f32_16x16x32_bf16(
                      false, a, false, b, (short)0, acc, false, false);
        }
        const int mb = (tm << 4) + hi8, n = cb + lo16;
#pragma unroll
        for (int r = 0; r < 8; ++r)
            og[(mb + r) * DIM + n] = acc[r];
    }
}

// ---------------------------------------------------------------------------
extern "C" void kernel_launch(void* const* d_in, const int* in_sizes, int n_in,
                              void* d_out, int out_size, void* d_ws, size_t ws_size,
                              hipStream_t stream) {
    const float* x         = (const float*)d_in[0];
    const float* theta_max = (const float*)d_in[1];
    const float* qkv_w     = (const float*)d_in[2];
    const float* qkv_b     = (const float*)d_in[3];
    const float* proj_w    = (const float*)d_in[4];
    const float* proj_b    = (const float*)d_in[5];
    const float* a_p       = (const float*)d_in[6];
    const float* b_p       = (const float*)d_in[7];
    const float* a_r       = (const float*)d_in[8];
    const float* b_r       = (const float*)d_in[9];
    float* out = (float*)d_out;

    char* ws = (char*)d_ws;
    __bf16* wqkv  = (__bf16*)(ws + WS_WQKV);
    __bf16* wproj = (__bf16*)(ws + WS_WPROJ);
    float*  bias  = (float*)(ws + WS_BIAS);

    // (768*256 + 256*256) = 262144 elements -> 1024 blocks
    prep_weights_kernel<<<1024, 256, 0, stream>>>(qkv_w, proj_w, wqkv, wproj);
    prep_bias_kernel<<<16, 256, 0, stream>>>(theta_max, a_p, b_p, a_r, b_r, bias);
    attn_kernel<<<2048, 256, SMEM_BYTES, stream>>>(x, wqkv, qkv_b, wproj,
                                                   proj_b, bias, out);
}
